// KoLeoLoss_2808908612227
// MI455X (gfx1250) — matrix operand order repaired and checked by reading, not compiled, not run on hardware
//
#include <hip/hip_runtime.h>
#include <hip/hip_bf16.h>

typedef __attribute__((ext_vector_type(16))) _Float16 v16h;
typedef __attribute__((ext_vector_type(8)))  _Float16 v8h;
typedef __attribute__((ext_vector_type(8)))  float    v8f;

#define N_ROWS 16384
#define DIM    512
#define KEPS   1e-8f

#define COL_SPLIT       8
#define COLS_PER_BLOCK  (N_ROWS / COL_SPLIT)     // 2048
#define TILES_PER_BLOCK (COLS_PER_BLOCK / 16)    // 128
#define TILE_BYTES      (16 * DIM * 2)           // 16 KB

// ---------------------------------------------------------------------------
// Phase 1: L2-normalize each row of the fp32 input, emit f16 copy for WMMA.
// One wave (32 lanes) per row; lane handles 16 contiguous elements.
// ---------------------------------------------------------------------------
__global__ __launch_bounds__(256) void koleo_normalize_f16(
    const float* __restrict__ in, _Float16* __restrict__ xh) {
  const int wave = threadIdx.x >> 5;
  const int lane = threadIdx.x & 31;
  const int row  = blockIdx.x * 8 + wave;
  const float* src = in + (size_t)row * DIM + lane * 16;

  float4 a0 = ((const float4*)src)[0];
  float4 a1 = ((const float4*)src)[1];
  float4 a2 = ((const float4*)src)[2];
  float4 a3 = ((const float4*)src)[3];

  float s = a0.x*a0.x + a0.y*a0.y + a0.z*a0.z + a0.w*a0.w
          + a1.x*a1.x + a1.y*a1.y + a1.z*a1.z + a1.w*a1.w
          + a2.x*a2.x + a2.y*a2.y + a2.z*a2.z + a2.w*a2.w
          + a3.x*a3.x + a3.y*a3.y + a3.z*a3.z + a3.w*a3.w;
  #pragma unroll
  for (int off = 16; off > 0; off >>= 1) s += __shfl_xor(s, off, 32);

  const float scale = 1.0f / fmaxf(sqrtf(s), KEPS);

  v16h h;
  h[0]  = (_Float16)(a0.x*scale); h[1]  = (_Float16)(a0.y*scale);
  h[2]  = (_Float16)(a0.z*scale); h[3]  = (_Float16)(a0.w*scale);
  h[4]  = (_Float16)(a1.x*scale); h[5]  = (_Float16)(a1.y*scale);
  h[6]  = (_Float16)(a1.z*scale); h[7]  = (_Float16)(a1.w*scale);
  h[8]  = (_Float16)(a2.x*scale); h[9]  = (_Float16)(a2.y*scale);
  h[10] = (_Float16)(a2.z*scale); h[11] = (_Float16)(a2.w*scale);
  h[12] = (_Float16)(a3.x*scale); h[13] = (_Float16)(a3.y*scale);
  h[14] = (_Float16)(a3.z*scale); h[15] = (_Float16)(a3.w*scale);

  *(v16h*)(xh + (size_t)row * DIM + lane * 16) = h;
}

// ---------------------------------------------------------------------------
// Phase 2: Gram matrix dots = x . x^T via v_wmma_f32_16x16x32_f16, fused with
// diagonal masking and per-row argmax over a 1/8 column slice.
//   grid = (64 row-strips, 8 column splits), block = 8 waves.
//   Two 16-row A tiles per wave resident in VGPRs (af[2][16] = 256 VGPR) so
//   every B fragment from LDS feeds two WMMAs (1 ds_load : 1 wmma).
//   B tiles double-buffered in LDS via global_load_async_to_lds_b128 (no
//   data VGPRs, ASYNCcnt-tracked), one barrier per tile.
// ---------------------------------------------------------------------------
__global__ __launch_bounds__(256) void koleo_gram_argmax(
    const _Float16* __restrict__ xh,
    float* __restrict__ pV, int* __restrict__ pI) {
  __shared__ _Float16 bl[2][16 * DIM];          // 2 x 16 KB B tiles

  const int tid   = threadIdx.x;
  const int wave  = tid >> 5;
  const int lane  = tid & 31;
  const int half  = lane >> 4;                  // 0: lanes 0-15, 1: lanes 16-31
  const int l16   = lane & 15;
  const int rbase = blockIdx.x * 256 + wave * 32;        // 2 row tiles / wave
  const int cbase = blockIdx.y * COLS_PER_BLOCK;

  // Global source of this block's column slice; per-thread 4 x 16B per tile.
  const char* gbase = (const char*)(xh + (size_t)cbase * DIM) + tid * 16;
  // LDS byte offsets (flat-LDS aperture: low 32 bits == LDS offset).
  const unsigned ldsA = (unsigned)(uintptr_t)(&bl[0][0]) + tid * 16;
  const unsigned ldsB = (unsigned)(uintptr_t)(&bl[1][0]) + tid * 16;

  // Kick off tile 0 into buffer 0 (async DMA, no data VGPRs).
  {
    unsigned long long g = (unsigned long long)(uintptr_t)gbase;
    #pragma unroll
    for (int q = 0; q < 4; ++q) {
      asm volatile("global_load_async_to_lds_b128 %0, %1, off"
                   :: "v"(ldsA + q * 4096),
                      "v"(g + (unsigned long long)(q * 4096))
                   : "memory");
    }
  }

  // Preload both A tiles (32 rows x K=512) into registers.
  // Layout (16-bit A 16x32, matches B): lane m -> K kk*32+0..15,
  //                                     lane m+16 -> K kk*32+16..31.
  v16h af[2][16];
  #pragma unroll
  for (int t = 0; t < 2; ++t) {
    const _Float16* rp = xh + (size_t)(rbase + t * 16 + l16) * DIM;
    #pragma unroll
    for (int kk = 0; kk < 16; ++kk) {
      const int kb = kk * 32 + (half ? 16 : 0);
      v8h lo = *(const v8h*)(rp + kb);
      v8h hi = *(const v8h*)(rp + kb + 8);
      v16h a;
      #pragma unroll
      for (int q = 0; q < 8; ++q) { a[q] = lo[q]; a[8 + q] = hi[q]; }
      af[t][kk] = a;
    }
  }

  float bestV[2][8];
  int   bestI[2][8];
  #pragma unroll
  for (int t = 0; t < 2; ++t)
    #pragma unroll
    for (int r = 0; r < 8; ++r) { bestV[t][r] = -2.0f; bestI[t][r] = 0; }

  for (int it = 0; it < TILES_PER_BLOCK; ++it) {
    // My async loads for tile `it` are in LDS; barrier makes everyone's visible.
    asm volatile("s_wait_asynccnt 0" ::: "memory");
    __syncthreads();

    // Prefetch tile it+1 into the other buffer. Issued only after the
    // barrier, so the last reader of that buffer (compute it-1) is done.
    if (it + 1 < TILES_PER_BLOCK) {
      unsigned long long g = (unsigned long long)(uintptr_t)gbase
                           + (unsigned long long)(it + 1) * TILE_BYTES;
      const unsigned l = (it & 1) ? ldsA : ldsB;
      #pragma unroll
      for (int q = 0; q < 4; ++q) {
        asm volatile("global_load_async_to_lds_b128 %0, %1, off"
                     :: "v"(l + q * 4096),
                        "v"(g + (unsigned long long)(q * 4096))
                     : "memory");
      }
    }

    const _Float16* bb = bl[it & 1];
    v8f c0 = {};
    v8f c1 = {};
    #pragma unroll
    for (int kk = 0; kk < 16; ++kk) {
      // B (16-bit 32x16): lane n -> K kk*32+0..15, lane n+16 -> K +16..31.
      const v16h b = *(const v16h*)(bb + l16 * DIM + kk * 32 + half * 16);
      c0 = __builtin_amdgcn_wmma_f32_16x16x32_f16(
               false, af[0][kk], false, b, (short)0, c0, false, false);
      c1 = __builtin_amdgcn_wmma_f32_16x16x32_f16(
               false, af[1][kk], false, b, (short)0, c1, false, false);
    }

    // C layout: VGPR r = rows {tbase+r (lanes 0-15), tbase+r+8 (lanes 16-31)},
    // column = cbase + it*16 + l16. Mask diagonal, update running argmax.
    const int col = cbase + it * 16 + l16;
    #pragma unroll
    for (int r = 0; r < 8; ++r) {
      const int g0 = rbase + r + half * 8;
      const int g1 = g0 + 16;
      float v0 = c0[r];
      float v1 = c1[r];
      if (col == g0) v0 = -2.0f;
      if (col == g1) v1 = -2.0f;
      if (v0 > bestV[0][r]) { bestV[0][r] = v0; bestI[0][r] = col; }
      if (v1 > bestV[1][r]) { bestV[1][r] = v1; bestI[1][r] = col; }
    }
  }

  // Cross-lane argmax within each 16-lane half (lower index wins ties),
  // then write this column-slice's partial result.
  #pragma unroll
  for (int t = 0; t < 2; ++t) {
    #pragma unroll
    for (int r = 0; r < 8; ++r) {
      float v = bestV[t][r];
      int   i = bestI[t][r];
      #pragma unroll
      for (int off = 1; off < 16; off <<= 1) {
        float ov = __shfl_xor(v, off, 32);
        int   oi = __shfl_xor(i, off, 32);
        if (ov > v || (ov == v && oi < i)) { v = ov; i = oi; }
      }
      bestV[t][r] = v; bestI[t][r] = i;
    }
    if (l16 == 0) {
      #pragma unroll
      for (int r = 0; r < 8; ++r) {
        const int row = rbase + t * 16 + half * 8 + r;
        pV[(size_t)row * COL_SPLIT + blockIdx.y] = bestV[t][r];
        pI[(size_t)row * COL_SPLIT + blockIdx.y] = bestI[t][r];
      }
    }
  }
}

// ---------------------------------------------------------------------------
// Phase 2b: merge the COL_SPLIT partial argmax results per row.
// Ascending split order + strict '>' preserves lowest-index tie-break.
// ---------------------------------------------------------------------------
__global__ __launch_bounds__(256) void koleo_argmax_merge(
    const float* __restrict__ pV, const int* __restrict__ pI,
    int* __restrict__ I) {
  const int row = blockIdx.x * 256 + threadIdx.x;
  float bv = -3.0f;
  int   bi = 0;
  #pragma unroll
  for (int p = 0; p < COL_SPLIT; ++p) {
    const float v = pV[(size_t)row * COL_SPLIT + p];
    const int   i = pI[(size_t)row * COL_SPLIT + p];
    if (v > bv) { bv = v; bi = i; }
  }
  I[row] = bi;
}

// ---------------------------------------------------------------------------
// Phase 3: fp32 distances. Renormalize rows i and I[i] from the ORIGINAL
// input (full fp32 precision), term[i] = log(||x_i - x_j + eps|| + eps).
// One wave per row.
// ---------------------------------------------------------------------------
__global__ __launch_bounds__(256) void koleo_dist(
    const float* __restrict__ in, const int* __restrict__ I,
    float* __restrict__ terms) {
  const int wave = threadIdx.x >> 5;
  const int lane = threadIdx.x & 31;
  const int row  = blockIdx.x * 8 + wave;
  const int j    = I[row];

  const float* a = in + (size_t)row * DIM + lane * 16;
  const float* b = in + (size_t)j   * DIM + lane * 16;

  float av[16], bv[16];
  float sa = 0.0f, sb = 0.0f;
  #pragma unroll
  for (int t = 0; t < 4; ++t) {
    float4 fa = ((const float4*)a)[t];
    float4 fb = ((const float4*)b)[t];
    av[4*t+0] = fa.x; av[4*t+1] = fa.y; av[4*t+2] = fa.z; av[4*t+3] = fa.w;
    bv[4*t+0] = fb.x; bv[4*t+1] = fb.y; bv[4*t+2] = fb.z; bv[4*t+3] = fb.w;
    sa += fa.x*fa.x + fa.y*fa.y + fa.z*fa.z + fa.w*fa.w;
    sb += fb.x*fb.x + fb.y*fb.y + fb.z*fb.z + fb.w*fb.w;
  }
  #pragma unroll
  for (int off = 16; off > 0; off >>= 1) {
    sa += __shfl_xor(sa, off, 32);
    sb += __shfl_xor(sb, off, 32);
  }
  const float ia = 1.0f / fmaxf(sqrtf(sa), KEPS);
  const float ib = 1.0f / fmaxf(sqrtf(sb), KEPS);

  float d2 = 0.0f;
  #pragma unroll
  for (int t = 0; t < 16; ++t) {
    const float d = av[t] * ia - bv[t] * ib + KEPS;
    d2 += d * d;
  }
  #pragma unroll
  for (int off = 16; off > 0; off >>= 1) d2 += __shfl_xor(d2, off, 32);

  if (lane == 0) terms[row] = logf(sqrtf(d2) + KEPS);
}

// ---------------------------------------------------------------------------
// Phase 4: loss = -mean(terms). Single block reduction.
// ---------------------------------------------------------------------------
__global__ __launch_bounds__(256) void koleo_reduce(
    const float* __restrict__ terms, float* __restrict__ out) {
  __shared__ float sh[8];
  float s = 0.0f;
  for (int i = threadIdx.x; i < N_ROWS; i += 256) s += terms[i];
  #pragma unroll
  for (int off = 16; off > 0; off >>= 1) s += __shfl_xor(s, off, 32);
  if ((threadIdx.x & 31) == 0) sh[threadIdx.x >> 5] = s;
  __syncthreads();
  if (threadIdx.x == 0) {
    float tot = 0.0f;
    #pragma unroll
    for (int w = 0; w < 8; ++w) tot += sh[w];
    out[0] = -tot / (float)N_ROWS;
  }
}

// ---------------------------------------------------------------------------
extern "C" void kernel_launch(void* const* d_in, const int* in_sizes, int n_in,
                              void* d_out, int out_size, void* d_ws, size_t ws_size,
                              hipStream_t stream) {
  const float* in = (const float*)d_in[0];
  char* ws = (char*)d_ws;

  size_t off = 0;
  _Float16* xh    = (_Float16*)(ws + off); off += (size_t)N_ROWS * DIM * 2;  // 16 MB
  int*      I     = (int*)     (ws + off); off += (size_t)N_ROWS * 4;        // 64 KB
  float*    terms = (float*)   (ws + off); off += (size_t)N_ROWS * 4;        // 64 KB
  float*    pV    = (float*)   (ws + off); off += (size_t)N_ROWS * COL_SPLIT * 4;
  int*      pI    = (int*)     (ws + off);                                   // 512 KB ea
  float* out = (float*)d_out;

  koleo_normalize_f16<<<N_ROWS / 8, 256, 0, stream>>>(in, xh);
  dim3 g2(N_ROWS / 256, COL_SPLIT);
  koleo_gram_argmax  <<<g2,          256, 0, stream>>>(xh, pV, pI);
  koleo_argmax_merge <<<N_ROWS/256,  256, 0, stream>>>(pV, pI, I);
  koleo_dist         <<<N_ROWS / 8,  256, 0, stream>>>(in, I, terms);
  koleo_reduce       <<<1,           256, 0, stream>>>(terms, out);
}
